// HyperGNN_1331439862294
// MI455X (gfx1250) — matrix operand order repaired and checked by reading, not compiled, not run on hardware
//
#include <hip/hip_runtime.h>
#include <hip/hip_bf16.h>

typedef __attribute__((ext_vector_type(16))) _Float16 v16h;
typedef __attribute__((ext_vector_type(8)))  float    v8f;
typedef __attribute__((ext_vector_type(4)))  unsigned int v4u;
typedef __attribute__((ext_vector_type(8)))  int      i32x8;
typedef __attribute__((ext_vector_type(4)))  int      i32x4;

#define NN   100000
#define NNZV 600000
#define HIDF 128

#if defined(__has_builtin)
#if __has_builtin(__builtin_amdgcn_tensor_load_to_lds) && __has_builtin(__builtin_amdgcn_s_wait_tensorcnt)
#define HAVE_TDM 1
#endif
#endif
#ifndef HAVE_TDM
#define HAVE_TDM 0
#endif

// ---------------------------------------------------------------------------
// f32 -> f16 convert
__global__ void k_f32_to_f16(const float* __restrict__ in, _Float16* __restrict__ out, int n) {
    int i = blockIdx.x * blockDim.x + threadIdx.x;
    if (i < n) out[i] = (_Float16)in[i];
}

// W [K,N] f32 -> WT [Npad,K] f16 (zero pad cols >= N)
__global__ void k_transpose_w(const float* __restrict__ W, _Float16* __restrict__ WT,
                              int K, int N, int Npad) {
    int i = blockIdx.x * blockDim.x + threadIdx.x;
    if (i >= Npad * K) return;
    int n = i / K, k = i % K;
    WT[(size_t)n * K + k] = (n < N) ? (_Float16)W[(size_t)k * N + n] : (_Float16)0.0f;
}

// node degree D and hyperedge size B via float atomics
__global__ void k_degrees(const long long* __restrict__ nidx, const long long* __restrict__ eidx,
                          float* __restrict__ D, float* __restrict__ B, int nnz) {
    int i = blockIdx.x * blockDim.x + threadIdx.x;
    if (i >= nnz) return;
    atomicAdd(&D[(int)nidx[i]], 1.0f);
    atomicAdd(&B[(int)eidx[i]], 1.0f);
}

__global__ void k_invert(float* __restrict__ D, float* __restrict__ B, int n) {
    int i = blockIdx.x * blockDim.x + threadIdx.x;
    if (i >= n) return;
    float d = D[i]; D[i] = (d > 0.0f) ? 1.0f / d : 0.0f;
    float b = B[i]; B[i] = (b > 0.0f) ? 1.0f / b : 0.0f;
}

// E[edge] += Y[node]  (one wave per nnz, float4 per lane, 4 atomics)
__global__ __launch_bounds__(256)
void k_scatter_n2e(const float* __restrict__ Y, const long long* __restrict__ nidx,
                   const long long* __restrict__ eidx, float* __restrict__ E, int nnz) {
    int gw = blockIdx.x * 8 + (threadIdx.x >> 5);
    if (gw >= nnz) return;
    int lane = threadIdx.x & 31;
    int node = (int)nidx[gw], edge = (int)eidx[gw];
    float4 v = ((const float4*)(Y + (size_t)node * HIDF))[lane];
    float* dst = E + (size_t)edge * HIDF + lane * 4;
    atomicAdd(dst + 0, v.x); atomicAdd(dst + 1, v.y);
    atomicAdd(dst + 2, v.z); atomicAdd(dst + 3, v.w);
}

// O[node] += E[edge] * Binv[edge]
__global__ __launch_bounds__(256)
void k_scatter_e2n(const float* __restrict__ E, const float* __restrict__ Binv,
                   const long long* __restrict__ nidx, const long long* __restrict__ eidx,
                   float* __restrict__ O, int nnz) {
    int gw = blockIdx.x * 8 + (threadIdx.x >> 5);
    if (gw >= nnz) return;
    int lane = threadIdx.x & 31;
    int node = (int)nidx[gw], edge = (int)eidx[gw];
    float s = Binv[edge];
    float4 v = ((const float4*)(E + (size_t)edge * HIDF))[lane];
    float* dst = O + (size_t)node * HIDF + lane * 4;
    atomicAdd(dst + 0, v.x * s); atomicAdd(dst + 1, v.y * s);
    atomicAdd(dst + 2, v.z * s); atomicAdd(dst + 3, v.w * s);
}

// per-feature sum / sumsq of u = O[n,f]*Dinv[n]
#define STAT_ROWS 250
__global__ __launch_bounds__(128)
void k_bn_stats(const float* __restrict__ O, const float* __restrict__ Dinv,
                float* __restrict__ Ssum, float* __restrict__ Ssq, int nrows) {
    int f  = threadIdx.x;
    int r0 = blockIdx.x * STAT_ROWS;
    int r1 = r0 + STAT_ROWS; if (r1 > nrows) r1 = nrows;
    float s = 0.0f, q = 0.0f;
    for (int r = r0; r < r1; ++r) {
        float u = O[(size_t)r * HIDF + f] * Dinv[r];
        s += u; q += u * u;
    }
    atomicAdd(&Ssum[f], s);
    atomicAdd(&Ssq[f], q);
}

__global__ void k_bn_params(const float* __restrict__ Ssum, const float* __restrict__ Ssq,
                            const float* __restrict__ g, const float* __restrict__ beta,
                            float* __restrict__ scale, float* __restrict__ shift, float invN) {
    int f = threadIdx.x;
    float m   = Ssum[f] * invN;
    float var = Ssq[f] * invN - m * m;
    float inv = rsqrtf(var + 1e-5f);
    float sc  = g[f] * inv;
    scale[f] = sc;
    shift[f] = beta[f] - m * sc;
}

// h = relu(BN(O*Dinv)) -> f16 activations for next GEMM
__global__ void k_bn_apply(const float* __restrict__ O, const float* __restrict__ Dinv,
                           const float* __restrict__ scale, const float* __restrict__ shift,
                           _Float16* __restrict__ Xh, int total) {
    int i = blockIdx.x * blockDim.x + threadIdx.x;
    if (i >= total) return;
    int n = i >> 7, f = i & 127;
    float u = O[i] * Dinv[n];
    float v = fmaf(u, scale[f], shift[f]);
    v = v > 0.0f ? v : 0.0f;
    Xh[i] = (_Float16)v;
}

// ---------------------------------------------------------------------------
// WMMA GEMM: C[M,N] = A[M,K](f16) * BT[N,K](f16)^T, fp32 accumulate.
// 256 threads = 8 waves; each wave owns a 16 x (NT*16) strip.
// Weight tile [N][KC] (KC<=128, <=32KB) staged into LDS by the Tensor Data
// Mover (one descriptor issued by wave 0, tracked with TENSORcnt); manual
// v4u copy loop as fallback when the TDM builtin is unavailable.
template<int NT, bool RELU, bool OUT16>
__global__ __launch_bounds__(256)
void k_gemm_wmma(const _Float16* __restrict__ A, const _Float16* __restrict__ BT,
                 const float* __restrict__ bias, float* __restrict__ Cf,
                 _Float16* __restrict__ Ch, int M, int K, int storeCols, int ldOut) {
    extern __shared__ _Float16 smem[];
    const int N   = NT * 16;
    const int KC  = (K < 128) ? K : 128;
    const int tid = threadIdx.x;
    const int lane = tid & 31;
    const int hi   = lane >> 4;          // half-wave select
    const int tile = blockIdx.x * 8 + (tid >> 5);
    const bool active = (tile * 16) < M;
    const int row0 = tile * 16;
    int arow = row0 + (lane & 15);
    if (arow >= M) arow = M - 1;

    v8f acc[NT];
#pragma unroll
    for (int j = 0; j < NT; ++j)
#pragma unroll
        for (int e = 0; e < 8; ++e) acc[j][e] = 0.0f;

    for (int kc = 0; kc < K; kc += KC) {
#if HAVE_TDM
        if ((tid >> 5) == 0) {           // wave 0 drives the TDM
            unsigned ldsOff = (unsigned)(size_t)(void*)smem;
            unsigned long long ga =
                (unsigned long long)(size_t)(const void*)BT + (unsigned long long)kc * 2ull;
            unsigned kcu = (unsigned)KC, nu = (unsigned)N, ku = (unsigned)K;
            // D# group 0: count=1 | lds_addr | global_addr(57b) | type=2
            v4u g0;
            g0[0] = 1u;
            g0[1] = ldsOff;
            g0[2] = (unsigned)(ga & 0xffffffffull);
            g0[3] = (unsigned)((ga >> 32) & 0x1ffffffull) | (2u << 30);
            // D# group 1: data_size=2B, tensor_dim0=KC, tensor_dim1=N,
            //             tile_dim0=KC, tile_dim1=N, tensor_dim0_stride=K
            i32x8 g1;
            g1[0] = (int)(1u << 16);                      // data_size code 1 (2 bytes)
            g1[1] = (int)(kcu << 16);                     // tensor_dim0[15:0]
            g1[2] = (int)((kcu >> 16) | (nu << 16));      // tensor_dim0[31:16] | tensor_dim1[15:0]
            g1[3] = (int)((nu >> 16) | (kcu << 16));      // tensor_dim1[31:16] | tile_dim0
            g1[4] = (int)nu;                              // tile_dim1 (tile_dim2 = 0)
            g1[5] = (int)ku;                              // tensor_dim0_stride[31:0] (elements)
            g1[6] = 0;
            g1[7] = 0;
            i32x4 z4; z4[0] = 0; z4[1] = 0; z4[2] = 0; z4[3] = 0;
#if __clang_major__ >= 23
            i32x8 z8;
#pragma unroll
            for (int q = 0; q < 8; ++q) z8[q] = 0;
            __builtin_amdgcn_tensor_load_to_lds(g0, g1, z4, z4, z8, 0);
#else
            __builtin_amdgcn_tensor_load_to_lds(g0, g1, z4, z4, 0);
#endif
            __builtin_amdgcn_s_wait_tensorcnt(0);
        }
        __syncthreads();
#else
        const int rowChunks = KC >> 3;   // 16B chunks per staged row
        for (int c = tid; c < N * rowChunks; c += 256) {
            int n = c / rowChunks, jj = c - n * rowChunks;
            *(v4u*)(smem + n * KC + jj * 8) =
                *(const v4u*)(BT + (size_t)n * K + kc + jj * 8);
        }
        __syncthreads();
#endif
        if (active) {
            const _Float16* ap = A + (size_t)arow * K + kc + hi * 8;
            for (int kb = 0; kb < KC; kb += 32) {
                union { v16h h; v4u u[2]; } af;
                af.u[0] = *(const v4u*)(ap + kb);        // K = klo+0..7
                af.u[1] = *(const v4u*)(ap + kb + 16);   // K = klo+16..23
#pragma unroll
                for (int j = 0; j < NT; ++j) {
                    union { v16h h; v4u u[2]; } bf;
                    const _Float16* bp = smem + (j * 16 + (lane & 15)) * KC + kb + hi * 8;
                    bf.u[0] = *(const v4u*)bp;
                    bf.u[1] = *(const v4u*)(bp + 16);
                    acc[j] = __builtin_amdgcn_wmma_f32_16x16x32_f16(
                        false, af.h, false, bf.h, (short)0, acc[j], false, false);
                }
            }
        }
        __syncthreads();
    }
    if (!active) return;
#pragma unroll
    for (int j = 0; j < NT; ++j) {
        int col = j * 16 + (lane & 15);
        if (col >= storeCols) continue;
        float bv = bias ? bias[col] : 0.0f;
#pragma unroll
        for (int e = 0; e < 8; ++e) {
            int row = row0 + hi * 8 + e;             // C/D: lanes>=16 hold M+8
            if (row >= M) continue;
            float v = acc[j][e] + bv;
            if (RELU) v = v > 0.0f ? v : 0.0f;
            if (OUT16) Ch[(size_t)row * ldOut + col] = (_Float16)v;
            else       Cf[(size_t)row * ldOut + col] = v;
        }
    }
}

// ---------------------------------------------------------------------------
extern "C" void kernel_launch(void* const* d_in, const int* in_sizes, int n_in,
                              void* d_out, int out_size, void* d_ws, size_t ws_size,
                              hipStream_t stream) {
    const float*     x   = (const float*)d_in[0];
    const long long* he  = (const long long*)d_in[1];   // [2, NNZ] int64
    const float* W1  = (const float*)d_in[2];
    const float* W2  = (const float*)d_in[4];
    const float* W3  = (const float*)d_in[6];
    const float* g1  = (const float*)d_in[8],  *bt1 = (const float*)d_in[9];
    const float* g2  = (const float*)d_in[10], *bt2 = (const float*)d_in[11];
    const float* g3  = (const float*)d_in[12], *bt3 = (const float*)d_in[13];
    const float* Wc1 = (const float*)d_in[14], *bc1 = (const float*)d_in[15];
    const float* Wc2 = (const float*)d_in[16], *bc2 = (const float*)d_in[17];
    const long long* nidx = he;
    const long long* eidx = he + NNZV;

    char* ws = (char*)d_ws; size_t off = 0;
    auto alloc = [&](size_t bytes) -> void* {
        void* p = ws + off; off += (bytes + 255) & ~(size_t)255; return p;
    };
    _Float16* Xh   = (_Float16*)alloc((size_t)NN * 256 * 2);
    float*    Y    = (float*)   alloc((size_t)NN * HIDF * 4);   // O aliases Y
    float*    E    = (float*)   alloc((size_t)NN * HIDF * 4);
    _Float16* C1h  = (_Float16*)alloc((size_t)NN * 64 * 2);
    float*    Dinv = (float*)   alloc((size_t)NN * 4);
    float*    Binv = (float*)   alloc((size_t)NN * 4);
    float*    Ssum = (float*)   alloc(512);
    float*    Ssq  = (float*)   alloc(512);
    float*    Scal = (float*)   alloc(512);
    float*    Shft = (float*)   alloc(512);
    _Float16* W1T  = (_Float16*)alloc(128 * 256 * 2);
    _Float16* W2T  = (_Float16*)alloc(128 * 128 * 2);
    _Float16* W3T  = (_Float16*)alloc(128 * 128 * 2);
    _Float16* Wc1T = (_Float16*)alloc(64 * 128 * 2);
    _Float16* Wc2T = (_Float16*)alloc(16 * 64 * 2);
    float*    O    = Y;

    // ---- one-time prep ----
    hipMemsetAsync(Dinv, 0, (size_t)NN * 4, stream);
    hipMemsetAsync(Binv, 0, (size_t)NN * 4, stream);
    k_degrees<<<(NNZV + 255) / 256, 256, 0, stream>>>(nidx, eidx, Dinv, Binv, NNZV);
    k_invert<<<(NN + 255) / 256, 256, 0, stream>>>(Dinv, Binv, NN);
    k_f32_to_f16<<<(NN * 256) / 256, 256, 0, stream>>>(x, Xh, NN * 256);
    k_transpose_w<<<(128 * 256 + 255) / 256, 256, 0, stream>>>(W1, W1T, 256, 128, 128);
    k_transpose_w<<<(128 * 128 + 255) / 256, 256, 0, stream>>>(W2, W2T, 128, 128, 128);
    k_transpose_w<<<(128 * 128 + 255) / 256, 256, 0, stream>>>(W3, W3T, 128, 128, 128);
    k_transpose_w<<<(64 * 128 + 255) / 256, 256, 0, stream>>>(Wc1, Wc1T, 128, 64, 64);
    k_transpose_w<<<(16 * 64 + 255) / 256, 256, 0, stream>>>(Wc2, Wc2T, 64, 10, 16);

    const int gemmBlocks = (NN / 16 + 7) / 8;   // 782

    auto layer = [&](int K, const _Float16* WT, const float* g, const float* bt) {
        // Y = Xh @ W   (wmma, fp32 accumulate)
        k_gemm_wmma<8, false, false><<<gemmBlocks, 256, 128 * ((K < 128) ? K : 128) * 2, stream>>>(
            Xh, WT, nullptr, Y, nullptr, NN, K, 128, 128);
        // E = segsum_edge(Y[node])
        hipMemsetAsync(E, 0, (size_t)NN * HIDF * 4, stream);
        k_scatter_n2e<<<(NNZV + 7) / 8, 256, 0, stream>>>(Y, nidx, eidx, E, NNZV);
        // O = segsum_node(E[edge] * Binv)
        hipMemsetAsync(O, 0, (size_t)NN * HIDF * 4, stream);
        k_scatter_e2n<<<(NNZV + 7) / 8, 256, 0, stream>>>(E, Binv, nidx, eidx, O, NNZV);
        // BN stats on u = O*Dinv (conv bias cancels inside BN)
        hipMemsetAsync(Ssum, 0, 512, stream);
        hipMemsetAsync(Ssq, 0, 512, stream);
        k_bn_stats<<<(NN + STAT_ROWS - 1) / STAT_ROWS, 128, 0, stream>>>(O, Dinv, Ssum, Ssq, NN);
        k_bn_params<<<1, 128, 0, stream>>>(Ssum, Ssq, g, bt, Scal, Shft, 1.0f / (float)NN);
        // Xh = f16(relu(BN(u)))
        k_bn_apply<<<(NN * HIDF) / 256, 256, 0, stream>>>(O, Dinv, Scal, Shft, Xh, NN * HIDF);
    };

    layer(256, W1T, g1, bt1);
    layer(128, W2T, g2, bt2);
    layer(128, W3T, g3, bt3);

    // classifier: relu(h @ Wc1 + bc1) -> f16, then @ Wc2 + bc2 -> d_out (10 cols of padded 16)
    k_gemm_wmma<4, true, true><<<gemmBlocks, 256, 64 * 128 * 2, stream>>>(
        Xh, Wc1T, bc1, nullptr, C1h, NN, 128, 64, 64);
    k_gemm_wmma<1, false, false><<<gemmBlocks, 256, 16 * 64 * 2, stream>>>(
        C1h, Wc2T, bc2, (float*)d_out, nullptr, NN, 64, 10, 10);
}
